// MultiLayerFastLocalGraphModelV2_51170240365268
// MI455X (gfx1250) — matrix operand order repaired and checked by reading, not compiled, not run on hardware
//
#include <hip/hip_runtime.h>
#include <stdint.h>
#include <stddef.h>

// ---------------------------------------------------------------------------
// MultiLayerFastLocalGraphModelV2 forward on gfx1250 (MI455X).
// Dense layers: WMMA f16->f32 GEMM (v_wmma_f32_16x16x32_f16), wave32.
// A-tiles staged with GLOBAL_LOAD_ASYNC_TO_LDS_B128 (ASYNCcnt), double-
// buffered so the next K-slice streams while the current one is in the WMMAs.
// Each block computes a full 64 x Np stripe so A streams from HBM exactly once.
// Segment reductions via L2 atomics (42MB accumulator << 192MB L2).
// Requires ws_size >= ~1.6 GB.
// ---------------------------------------------------------------------------

typedef _Float16 h16;
typedef __attribute__((ext_vector_type(16))) _Float16 v16h;
typedef __attribute__((ext_vector_type(8)))  _Float16 v8h;
typedef __attribute__((ext_vector_type(8)))  float    v8f;

#define TPB 256
static const int S1 = 800000;
static const int K1 = 32768;
static const int E2 = 983040;

// ---------------- WMMA GEMM ----------------
// C = act(A @ B + bias) [+ Res]   act: 0=none, 1=relu, 2=leaky(0.01)
// A: (M x Kp) f16 row-major (Kp % 32 == 0); B: (Kp x Np) f16 row-major,
// Np == NB*32. Block tile 64(M) x Np(N); 8 wave32 waves (4m x 2n);
// each wave accumulates NB 16x16 tiles.
template <int NB>
__global__ __launch_bounds__(TPB) void k_gemm(
    const h16* __restrict__ A,
    const h16* __restrict__ B,
    const float* __restrict__ bias,
    h16* __restrict__ Ch,
    float* __restrict__ Cf, int ldcf, int ntrue,
    const h16* __restrict__ Res,
    int M, int Kp, int act)
{
  constexpr int Np = NB * 32;
  __shared__ h16 sA[2][64 * 32];
  __shared__ h16 sB[2][Np * 32];        // transposed: sB[buf][n*32 + k]
  const int tid  = threadIdx.x;
  const int m0   = blockIdx.x * 64;
  const int wave = tid >> 5;            // wave32
  const int lane = tid & 31;
  const int tm   = (wave >> 1) * 16;    // 0,16,32,48
  const int tnw  = (wave & 1) * 16;     // 0,16 within each 32-col chunk
  const int mr   = lane & 15;
  const int hi   = lane >> 4;

  v8f acc[NB];
#pragma unroll
  for (int nb = 0; nb < NB; ++nb) acc[nb] = (v8f){};

  // Per-thread A staging addresses (16B per thread per k-step).
  const int arow = tid >> 2, akq = (tid & 3) * 8;
  const h16* aptr = A + (size_t)(m0 + arow) * Kp + akq;
  // low 32 bits of the flat pointer == workgroup-relative LDS byte offset
  const unsigned sAd0 = (unsigned)(uintptr_t)(&sA[0][0] + arow * 32 + akq);
  const unsigned sAd1 = (unsigned)(uintptr_t)(&sA[1][0] + arow * 32 + akq);

  // B^T staging of one 32 x Np K-slice into sB[buf] (B is small / L2-hot).
  auto stageB = [&](int k0, int buf) {
    if (tid < 128) {
      int kk = tid >> 2, ng = (tid & 3) * 8;
#pragma unroll
      for (int nb = 0; nb < NB; ++nb) {
        v8h bv = *(const v8h*)(B + (size_t)(k0 + kk) * Np + nb * 32 + ng);
#pragma unroll
        for (int i = 0; i < 8; ++i)
          sB[buf][(nb * 32 + ng + i) * 32 + kk] = bv[i];
      }
    }
  };
  auto stageA = [&](int k0, int buf) {
    asm volatile("global_load_async_to_lds_b128 %0, %1, off"
                 :: "v"(buf ? sAd1 : sAd0), "v"(aptr + k0) : "memory");
  };

  // ---- prologue: stage slice 0 into buffer 0 ----
  stageA(0, 0);
  stageB(0, 0);
  asm volatile("s_wait_asynccnt 0x0" ::: "memory");
  __syncthreads();

  int cur = 0;
  for (int k0 = 0; k0 < Kp; k0 += 32) {
    const int nxt = cur ^ 1;
    if (k0 + 32 < Kp) {          // prefetch next K-slice into the other buffer
      stageA(k0 + 32, nxt);
      stageB(k0 + 32, nxt);
    }
    // A frag: lane holds M=mr, K = hi*8..hi*8+7 and 16+hi*8..16+hi*8+7
    union { v16h v; v8h h[2]; } af;
    const h16* ap = &sA[cur][0] + (tm + mr) * 32 + hi * 8;
    af.h[0] = *(const v8h*)ap;
    af.h[1] = *(const v8h*)(ap + 16);
#pragma unroll
    for (int nb = 0; nb < NB; ++nb) {
      // B frag: lane holds N=mr (in chunk nb), K = hi*8..hi*8+7 and 16+hi*8..16+hi*8+7
      union { v16h v; v8h h[2]; } bf;
      const h16* bp = &sB[cur][0] + (nb * 32 + tnw + mr) * 32 + hi * 8;
      bf.h[0] = *(const v8h*)bp;
      bf.h[1] = *(const v8h*)(bp + 16);
      acc[nb] = __builtin_amdgcn_wmma_f32_16x16x32_f16(
          false, af.v, false, bf.v, (short)0, acc[nb], false, false);
    }
    // single barrier per step: next-slice staging complete (asynccnt + dscnt)
    // AND everyone is done reading `cur` before it is overwritten.
    asm volatile("s_wait_asynccnt 0x0" ::: "memory");
    __syncthreads();
    cur = nxt;
  }
  // ---- epilogue: C/D layout n = lane&15, m = v + hi*8 ----
#pragma unroll
  for (int nb = 0; nb < NB; ++nb) {
    const int gn = nb * 32 + tnw + mr;
    const float bc = bias ? bias[gn] : 0.0f;
#pragma unroll
    for (int v = 0; v < 8; ++v) {
      int gm = m0 + tm + hi * 8 + v;
      float val = acc[nb][v] + bc;
      if (act == 1) val = val > 0.f ? val : 0.f;
      else if (act == 2) val = val > 0.f ? val : 0.01f * val;
      if (Res) val += (float)Res[(size_t)gm * Np + gn];
      if (Ch) Ch[(size_t)gm * Np + gn] = (h16)val;
      if (Cf && gn < ntrue) Cf[(size_t)gm * ldcf + gn] = val;
    }
  }
}

// ---------------- aux kernels ----------------
__global__ void k_pack_w(const float* __restrict__ W, h16* __restrict__ Wp,
                         int K, int N, int Kp, int Np) {
  int i = blockIdx.x * TPB + threadIdx.x;
  if (i >= Kp * Np) return;
  int k = i / Np, n = i % Np;
  float v = (k < K && n < N) ? W[k * N + n] : 0.f;
  Wp[i] = (h16)v;
}
__global__ void k_pack_b(const float* __restrict__ b, float* __restrict__ bp,
                         int N, int Np) {
  int i = blockIdx.x * TPB + threadIdx.x;
  if (i >= Np) return;
  bp[i] = (i < N) ? b[i] : 0.f;
}
__global__ void k_zero_f32(float* p, long long n) {
  long long i = (long long)blockIdx.x * TPB + threadIdx.x;
  if (i < n) p[i] = 0.f;
}
__global__ void k_f32_to_h16(const float* s, h16* d, long long n) {
  long long i = (long long)blockIdx.x * TPB + threadIdx.x;
  if (i < n) d[i] = (h16)s[i];
}
__global__ void k_copy_f32(const float* s, float* d, int n) {
  int i = blockIdx.x * TPB + threadIdx.x;
  if (i < n) d[i] = s[i];
}
__global__ void k_gather_kpc(const float* pc, const int* kidx, float* kpc, int K) {
  int k = blockIdx.x * TPB + threadIdx.x;
  if (k >= K) return;
  int p = kidx[k];
  kpc[3 * k + 0] = pc[3 * p + 0];
  kpc[3 * k + 1] = pc[3 * p + 1];
  kpc[3 * k + 2] = pc[3 * p + 2];
}
__global__ void k_build_x1(const float* pf, const float* pc, const float* kpc,
                           const int* setidx, h16* x1, int S) {
  int s = blockIdx.x * TPB + threadIdx.x;
  if (s >= S) return;
  int pi = setidx[2 * s], si = setidx[2 * s + 1];
  h16* r = x1 + (size_t)s * 32;
  r[0] = (h16)pf[pi];
  r[1] = (h16)(pc[3 * pi + 0] - kpc[3 * si + 0]);
  r[2] = (h16)(pc[3 * pi + 1] - kpc[3 * si + 1]);
  r[3] = (h16)(pc[3 * pi + 2] - kpc[3 * si + 2]);
#pragma unroll
  for (int j = 4; j < 32; ++j) r[j] = (h16)0.f;
}
// inputs are post-ReLU (>=0) so uint-bitcast atomicMax over a zeroed f32
// accumulator matches max(segment_max(x), 0).
__global__ void k_segmax8(const h16* X, const int* setidx, float* agg, long long nt) {
  long long i = (long long)blockIdx.x * TPB + threadIdx.x;
  if (i >= nt) return;
  long long s = i / 40;
  int c = (int)(i % 40);
  int seg = setidx[2 * s + 1];
  v8h x = *(const v8h*)(X + (size_t)s * 320 + c * 8);
  unsigned* dst = (unsigned*)(agg + (size_t)seg * 320 + c * 8);
#pragma unroll
  for (int j = 0; j < 8; ++j) atomicMax(dst + j, __float_as_uint((float)x[j]));
}
__global__ void k_segsum8(const h16* X, const int* edges, float* acc, long long nt) {
  long long i = (long long)blockIdx.x * TPB + threadIdx.x;
  if (i >= nt) return;
  long long e = i / 40;
  int c = (int)(i % 40);
  int dst = edges[2 * e + 1];
  v8h x = *(const v8h*)(X + (size_t)e * 320 + c * 8);
  float* d = acc + (size_t)dst * 320 + c * 8;
#pragma unroll
  for (int j = 0; j < 8; ++j) atomicAdd(d + j, (float)x[j]);
}
__global__ void k_deg(const int* edges, float* deg, int E) {
  int e = blockIdx.x * TPB + threadIdx.x;
  if (e < E) atomicAdd(deg + edges[2 * e + 1], 1.f);
}
__global__ void k_deg_fin(float* deg, int K) {
  int k = blockIdx.x * TPB + threadIdx.x;
  if (k < K) deg[k] = fmaxf(deg[k], 1.f);
}
__global__ void k_add_off(float* coords, const h16* off, int K) {
  int k = blockIdx.x * TPB + threadIdx.x;
  if (k >= K) return;
  coords[3 * k + 0] += (float)off[(size_t)k * 32 + 0];
  coords[3 * k + 1] += (float)off[(size_t)k * 32 + 1];
  coords[3 * k + 2] += (float)off[(size_t)k * 32 + 2];
}
__global__ void k_build_e8(const h16* H, const int* edges, h16* ein, long long nt) {
  long long i = (long long)blockIdx.x * TPB + threadIdx.x;
  if (i >= nt) return;
  long long e = i / 40;
  int c = (int)(i % 40);
  int src = edges[2 * e];
  *(v8h*)(ein + (size_t)e * 320 + c * 8) =
      *(const v8h*)(H + (size_t)src * 320 + c * 8);
}
__global__ void k_e_dc(const float* coords, const int* edges, h16* ein, int E) {
  int e = blockIdx.x * TPB + threadIdx.x;
  if (e >= E) return;
  int s = edges[2 * e], d = edges[2 * e + 1];
  h16* r = ein + (size_t)e * 320 + 300;
  r[0] = (h16)(coords[3 * d + 0] - coords[3 * s + 0]);
  r[1] = (h16)(coords[3 * d + 1] - coords[3 * s + 1]);
  r[2] = (h16)(coords[3 * d + 2] - coords[3 * s + 2]);
}
__global__ void k_mean(const float* esum, const float* deg, h16* out, long long n) {
  long long i = (long long)blockIdx.x * TPB + threadIdx.x;
  if (i < n) out[i] = (h16)(esum[i] / deg[i / 320]);
}
__global__ __launch_bounds__(TPB) void k_bn_stats(const h16* X, int ld, int M,
                                                  float* mean, float* rstd) {
  __shared__ float s1[TPB], s2[TPB];
  int f = blockIdx.x;
  float a = 0.f, b = 0.f;
  for (int m = threadIdx.x; m < M; m += TPB) {
    float x = (float)X[(size_t)m * ld + f];
    a += x; b += x * x;
  }
  s1[threadIdx.x] = a; s2[threadIdx.x] = b;
  __syncthreads();
  for (int off = TPB / 2; off > 0; off >>= 1) {
    if (threadIdx.x < off) {
      s1[threadIdx.x] += s1[threadIdx.x + off];
      s2[threadIdx.x] += s2[threadIdx.x + off];
    }
    __syncthreads();
  }
  if (threadIdx.x == 0) {
    float mu = s1[0] / M;
    float var = s2[0] / M - mu * mu;
    mean[f] = mu;
    rstd[f] = rsqrtf(fmaxf(var, 0.f) + 1e-5f);
  }
}
__global__ void k_bn_apply(h16* X, int ld, long long n, const float* mean,
                           const float* rstd) {
  long long i = (long long)blockIdx.x * TPB + threadIdx.x;
  if (i >= n) return;
  int f = (int)(i % ld);
  X[i] = (h16)(((float)X[i] - mean[f]) * rstd[f]);
}

// ---------------- host orchestration ----------------
struct PW { h16* W; float* b; int Kp, Np; };

extern "C" void kernel_launch(void* const* d_in, const int* in_sizes, int n_in,
                              void* d_out, int out_size, void* d_ws, size_t ws_size,
                              hipStream_t stream)
{
  (void)in_sizes; (void)n_in; (void)out_size; (void)ws_size;
  auto F  = [&](int i) { return (const float*)d_in[i]; };
  auto Ii = [&](int i) { return (const int*)d_in[i]; };

  // ---- input index map (setup_inputs insertion order; params = sorted pytree) ----
  const int P = 3; // params leaves start
  const int i_cls_lin1_W = P + 0, i_cls_lin1_b = P + 1, i_cls_Wo = P + 2,
            i_cls_Wv = P + 3, i_cls_bo = P + 4, i_cls_bv = P + 5,
            i_cls_outW = P + 6, i_cls_outb = P + 7;
  auto gbase = [&](int g) { return P + 8 + 6 * g; };   // edgeW,edgeb,offW,offb,updW,updb
  auto lbase = [&](int c) { return P + 26 + 14 * c; }; // lin1W,b,lin2W,b,mh1Wo,mh1Wv,mh1bo,mh1bv,mh2Wo,mh2Wv,mh2bo,mh2bv,outW,outb
  const int i_om_W0 = P + 82, i_om_b0 = P + 83, i_om_W1 = P + 84, i_om_b1 = P + 85;
  const int i_pm = P + 86; // W0,b0,W1,b1,W2,b2,W3,b3
  const int I_KIDX1 = 97, I_SET = 98, I_EDGES = 99;

  // ---- workspace carve (edge buffers aliased over dead point-phase buffers) ----
  char* base = (char*)d_ws;
  const size_t BIG = (size_t)E2 * 320 * sizeof(h16); // 629,145,600 B
  h16* BIGA = (h16*)base;         // a3 (S1x128) then e_in (E2x320)
  h16* BIGB = (h16*)(base + BIG); // a1/a4 then e_out
  char* p = base + 2 * BIG;
  auto alloc = [&](size_t bytes) {
    void* r = p; p += (bytes + 255) & ~(size_t)255; return r;
  };
  h16* a3    = BIGA;
  h16* e_in  = BIGA;
  h16* a1    = BIGB;                               // S1x32
  h16* a4    = BIGB;                               // S1x320 (a1 dead)
  h16* e_out = BIGB;                               // E2x320 (a4 dead)
  h16* a2    = (h16*)((char*)BIGB + 516000000ull); // S1x64, clear of a1/a4
  h16*  x1    = (h16*)alloc((size_t)S1 * 32 * 2);
  float* agg  = (float*)alloc((size_t)K1 * 320 * 4); // segmax / segsum acc
  h16*  h_a   = (h16*)alloc((size_t)K1 * 320 * 2);
  h16*  h_b   = (h16*)alloc((size_t)K1 * 320 * 2);
  h16*  meanh = (h16*)alloc((size_t)K1 * 320 * 2);
  h16*  t320a = (h16*)alloc((size_t)K1 * 320 * 2);
  h16*  t320b = (h16*)alloc((size_t)K1 * 320 * 2);
  h16*  t64a  = (h16*)alloc((size_t)K1 * 64 * 2);
  h16*  t64b  = (h16*)alloc((size_t)K1 * 64 * 2);
  h16*  t64c  = (h16*)alloc((size_t)K1 * 64 * 2);
  h16*  offb  = (h16*)alloc((size_t)K1 * 32 * 2);
  float* kpc  = (float*)alloc((size_t)K1 * 3 * 4);
  float* coords = (float*)alloc((size_t)K1 * 3 * 4);
  float* deg  = (float*)alloc((size_t)K1 * 4);
  float* bnm  = (float*)alloc(320 * 4);
  float* bnr  = (float*)alloc(320 * 4);

  auto packw = [&](int iw, int ib, int K, int N) {
    PW w; w.Kp = (K + 31) & ~31; w.Np = (N + 31) & ~31;
    w.W = (h16*)alloc((size_t)w.Kp * w.Np * 2);
    w.b = (float*)alloc((size_t)w.Np * 4);
    int nw = w.Kp * w.Np;
    k_pack_w<<<(nw + TPB - 1) / TPB, TPB, 0, stream>>>(F(iw), w.W, K, N, w.Kp, w.Np);
    k_pack_b<<<(w.Np + TPB - 1) / TPB, TPB, 0, stream>>>(F(ib), w.b, N, w.Np);
    return w;
  };
  auto gemm = [&](const h16* A, const PW& w, h16* Ch, float* Cf, int ldcf,
                  int ntrue, const h16* Res, int M, int act) {
    dim3 grid(M / 64);
    switch (w.Np / 32) {
      case 1:  k_gemm<1><<<grid, TPB, 0, stream>>>(A, w.W, w.b, Ch, Cf, ldcf, ntrue, Res, M, w.Kp, act); break;
      case 2:  k_gemm<2><<<grid, TPB, 0, stream>>>(A, w.W, w.b, Ch, Cf, ldcf, ntrue, Res, M, w.Kp, act); break;
      case 4:  k_gemm<4><<<grid, TPB, 0, stream>>>(A, w.W, w.b, Ch, Cf, ldcf, ntrue, Res, M, w.Kp, act); break;
      default: k_gemm<10><<<grid, TPB, 0, stream>>>(A, w.W, w.b, Ch, Cf, ldcf, ntrue, Res, M, w.Kp, act); break;
    }
  };
  auto bn = [&](h16* X, int Np, int M) {
    k_bn_stats<<<Np, TPB, 0, stream>>>(X, Np, M, bnm, bnr);
    long long n = (long long)M * Np;
    k_bn_apply<<<(int)((n + TPB - 1) / TPB), TPB, 0, stream>>>(X, Np, n, bnm, bnr);
  };
  auto zerof = [&](float* ptr, long long n) {
    k_zero_f32<<<(int)((n + TPB - 1) / TPB), TPB, 0, stream>>>(ptr, n);
  };

  // ---- pack weights (f32 -> padded f16) ----
  PW pm0 = packw(i_pm + 0, i_pm + 1, 4, 32);
  PW pm1 = packw(i_pm + 2, i_pm + 3, 32, 64);
  PW pm2 = packw(i_pm + 4, i_pm + 5, 64, 128);
  PW pm3 = packw(i_pm + 6, i_pm + 7, 128, 300);
  PW om0 = packw(i_om_W0, i_om_b0, 300, 300);
  PW om1 = packw(i_om_W1, i_om_b1, 300, 300);
  PW gE[3], gO[3], gU[3];
  for (int g = 0; g < 3; ++g) {
    int b = gbase(g);
    gE[g] = packw(b + 0, b + 1, 303, 300);
    gO[g] = packw(b + 2, b + 3, 300, 3);
    gU[g] = packw(b + 4, b + 5, 300, 300);
  }
  PW clsL  = packw(i_cls_lin1_W, i_cls_lin1_b, 300, 64);
  PW clsWv = packw(i_cls_Wv, i_cls_bv, 64, 64);
  PW clsWo = packw(i_cls_Wo, i_cls_bo, 64, 64);
  PW clsO  = packw(i_cls_outW, i_cls_outb, 64, 4);
  PW lo1[4], lo2[4], lWv1[4], lWo1[4], lWv2[4], lWo2[4], lOut[4];
  for (int c = 0; c < 4; ++c) {
    int b = lbase(c);
    lo1[c]  = packw(b + 0,  b + 1,  300, 300);
    lo2[c]  = packw(b + 2,  b + 3,  300, 64);
    lWo1[c] = packw(b + 4,  b + 6,  300, 300);
    lWv1[c] = packw(b + 5,  b + 7,  300, 300);
    lWo2[c] = packw(b + 8,  b + 10, 64, 64);
    lWv2[c] = packw(b + 9,  b + 11, 64, 64);
    lOut[c] = packw(b + 12, b + 13, 64, 7);
  }

  // ---- point phase: gather -> MLP(4,32,64,128,300) -> segment_max -> out_mlp ----
  k_gather_kpc<<<(K1 + TPB - 1) / TPB, TPB, 0, stream>>>(F(1), Ii(I_KIDX1), kpc, K1);
  k_build_x1<<<(S1 + TPB - 1) / TPB, TPB, 0, stream>>>(F(0), F(1), kpc, Ii(I_SET), x1, S1);
  gemm(x1, pm0, a1, nullptr, 0, 0, nullptr, S1, 1);
  gemm(a1, pm1, a2, nullptr, 0, 0, nullptr, S1, 1);
  gemm(a2, pm2, a3, nullptr, 0, 0, nullptr, S1, 1);
  gemm(a3, pm3, a4, nullptr, 0, 0, nullptr, S1, 1);
  zerof(agg, (long long)K1 * 320);
  long long nsm = (long long)S1 * 40;
  k_segmax8<<<(int)((nsm + TPB - 1) / TPB), TPB, 0, stream>>>(a4, Ii(I_SET), agg, nsm);
  long long nK = (long long)K1 * 320;
  k_f32_to_h16<<<(int)((nK + TPB - 1) / TPB), TPB, 0, stream>>>(agg, h_a, nK);
  gemm(h_a, om0, h_b, nullptr, 0, 0, nullptr, K1, 1);
  gemm(h_b, om1, h_a, nullptr, 0, 0, nullptr, K1, 1);
  h16* hc = h_a; h16* hn = h_b;

  // ---- degrees and coords ----
  zerof(deg, K1);
  k_deg<<<(E2 + TPB - 1) / TPB, TPB, 0, stream>>>(Ii(I_EDGES), deg, E2);
  k_deg_fin<<<(K1 + TPB - 1) / TPB, TPB, 0, stream>>>(deg, K1);
  k_copy_f32<<<(K1 * 3 + TPB - 1) / TPB, TPB, 0, stream>>>(F(2), coords, K1 * 3);

  // ---- 3 graph layers ----
  for (int g = 0; g < 3; ++g) {
    gemm(hc, gO[g], offb, nullptr, 0, 0, nullptr, K1, 1);
    k_add_off<<<(K1 + TPB - 1) / TPB, TPB, 0, stream>>>(coords, offb, K1);
    long long ne8 = (long long)E2 * 40;
    k_build_e8<<<(int)((ne8 + TPB - 1) / TPB), TPB, 0, stream>>>(hc, Ii(I_EDGES), e_in, ne8);
    k_e_dc<<<(E2 + TPB - 1) / TPB, TPB, 0, stream>>>(coords, Ii(I_EDGES), e_in, E2);
    gemm(e_in, gE[g], e_out, nullptr, 0, 0, nullptr, E2, 1);
    zerof(agg, nK);
    k_segsum8<<<(int)((ne8 + TPB - 1) / TPB), TPB, 0, stream>>>(e_out, Ii(I_EDGES), agg, ne8);
    k_mean<<<(int)((nK + TPB - 1) / TPB), TPB, 0, stream>>>(agg, deg, meanh, nK);
    gemm(meanh, gU[g], hn, nullptr, 0, 0, hc, K1, 1); // relu(update) + h
    h16* t = hc; hc = hn; hn = t;
  }

  float* out = (float*)d_out; // logits (K1 x 4), then boxes (K1 x 4 x 7)
  // ---- classification head ----
  gemm(hc, clsL, t64a, nullptr, 0, 0, nullptr, K1, 2);
  bn(t64a, 64, K1);
  gemm(t64a, clsWv, t64b, nullptr, 0, 0, nullptr, K1, 0);
  gemm(t64b, clsWo, t64c, nullptr, 0, 0, t64a, K1, 0);
  gemm(t64c, clsO, nullptr, out, 4, 4, nullptr, K1, 0);
  // ---- localization heads ----
  for (int c = 0; c < 4; ++c) {
    gemm(hc, lo1[c], t320a, nullptr, 0, 0, nullptr, K1, 2);
    bn(t320a, 320, K1);
    gemm(t320a, lWv1[c], t320b, nullptr, 0, 0, nullptr, K1, 0);
    gemm(t320b, lWo1[c], meanh, nullptr, 0, 0, t320a, K1, 0);
    gemm(meanh, lo2[c], t64a, nullptr, 0, 0, nullptr, K1, 2);
    bn(t64a, 64, K1);
    gemm(t64a, lWv2[c], t64b, nullptr, 0, 0, nullptr, K1, 0);
    gemm(t64b, lWo2[c], t64c, nullptr, 0, 0, t64a, K1, 0);
    gemm(t64c, lOut[c], nullptr, out + 131072 + c * 7, 28, 7, nullptr, K1, 0);
  }
}
